// TimeSeriesMamba_32315333935753
// MI455X (gfx1250) — compile-verified, matrix operand order checked by reading, not run on hardware
//
#include <hip/hip_runtime.h>
#include <hip/hip_bf16.h>

// ---------------------------------------------------------------------------
// TimeSeriesMamba forward on gfx1250 (MI455X).
//   B=4, L=512, D_MODEL=128, D_INNER=256, D_STATE=128, DT_RANK=8, D_HIDDEN=384
// GEMMs use v_wmma_f32_16x16x32_f16 (f16 inputs, f32 accumulate); each wave
// computes a 16x64 output strip (A-fragment reused across 4 WMMAs per K-step).
// Selective scan keeps state in VGPRs (fp32), one block per batch.
// ---------------------------------------------------------------------------

typedef __attribute__((ext_vector_type(16))) _Float16 v16h;
typedef __attribute__((ext_vector_type(8)))  float    v8f;

#define MTOK   2048   // B*L tokens
#define DMODEL 128
#define DINNER 256
#define DSTATE 128
#define DTRANK 8
#define DHID   384
#define LSEQ   512
#define XDBLW  264    // DT_RANK + 2*D_STATE

__device__ __forceinline__ float siluf(float x) { return x / (1.f + __expf(-x)); }

// Load one 16-element f16 fragment row: two contiguous 8-float runs at
// p[half*8 .. +7] and p[16+half*8 .. +7]  (ISA 16-bit A/B 16x32 layout).
__device__ __forceinline__ v16h load_frag(const float* __restrict__ p, int half) {
    const float4* q0 = (const float4*)(p + half * 8);
    const float4* q1 = (const float4*)(p + 16 + half * 8);
    float4 v0 = q0[0], v1 = q0[1], v2 = q1[0], v3 = q1[1];
    v16h r;
    r[0]=(_Float16)v0.x;  r[1]=(_Float16)v0.y;  r[2]=(_Float16)v0.z;  r[3]=(_Float16)v0.w;
    r[4]=(_Float16)v1.x;  r[5]=(_Float16)v1.y;  r[6]=(_Float16)v1.z;  r[7]=(_Float16)v1.w;
    r[8]=(_Float16)v2.x;  r[9]=(_Float16)v2.y;  r[10]=(_Float16)v2.z; r[11]=(_Float16)v2.w;
    r[12]=(_Float16)v3.x; r[13]=(_Float16)v3.y; r[14]=(_Float16)v3.z; r[15]=(_Float16)v3.w;
    return r;
}

// ---------------------------------------------------------------------------
// WMMA GEMM:  out[m][n] = sum_k A[m][k] * W[n][k]  (+ addend[m][n] if HAS_ADD)
// A: (2048, K) f32 row-major.  W: (N, K) f32 row-major.  out: (2048, N) f32.
// One wave per 16x64 strip (4 N-tiles); K must be a multiple of 32.
// A-fragment layout: lane m=L%16, half=L/16, elem e -> K = 16*(e/8)+8*half+(e%8).
// B-fragment mirrors A with column n = L%16 (W row n supplies the K run).
// C/D layout: lane L -> col n=L%16, VGPR r -> row M = r + 8*(L/16).
// ---------------------------------------------------------------------------
template<bool HAS_ADD>
__global__ __launch_bounds__(256)
void gemm_f16wmma(const float* __restrict__ A, const float* __restrict__ W,
                  const float* __restrict__ addend, float* __restrict__ out,
                  int Mtiles, int NGroups, int N, int K) {
    int wv   = (blockIdx.x * blockDim.x + threadIdx.x) >> 5;
    int lane = threadIdx.x & 31;
    if (wv >= Mtiles * NGroups) return;          // uniform per wave
    int tm   = wv / NGroups;
    int ng   = wv % NGroups;
    int half = lane >> 4;
    int l16  = lane & 15;
    int arow = tm * 16 + l16;
    const float* ap = A + (size_t)arow * K;

    int bcol[4];
    const float* wp[4];
    #pragma unroll
    for (int s = 0; s < 4; ++s) {
        int c   = (ng * 4 + s) * 16 + l16;
        bcol[s] = c;
        wp[s]   = W + (size_t)((c < N) ? c : 0) * K;   // keep EXEC uniform
    }

    v8f acc[4] = {};
    for (int kk = 0; kk < K; kk += 32) {
        v16h a  = load_frag(ap + kk, half);            // loaded once, used 4x
        v16h b0 = load_frag(wp[0] + kk, half);
        v16h b1 = load_frag(wp[1] + kk, half);
        v16h b2 = load_frag(wp[2] + kk, half);
        v16h b3 = load_frag(wp[3] + kk, half);
        acc[0] = __builtin_amdgcn_wmma_f32_16x16x32_f16(false, a, false, b0,
                                                        (short)0, acc[0], false, false);
        acc[1] = __builtin_amdgcn_wmma_f32_16x16x32_f16(false, a, false, b1,
                                                        (short)0, acc[1], false, false);
        acc[2] = __builtin_amdgcn_wmma_f32_16x16x32_f16(false, a, false, b2,
                                                        (short)0, acc[2], false, false);
        acc[3] = __builtin_amdgcn_wmma_f32_16x16x32_f16(false, a, false, b3,
                                                        (short)0, acc[3], false, false);
    }

    int rbase = tm * 16 + half * 8;
    #pragma unroll
    for (int s = 0; s < 4; ++s) {
        if (bcol[s] < N) {
            #pragma unroll
            for (int r = 0; r < 8; ++r) {
                size_t idx = (size_t)(rbase + r) * N + bcol[s];
                float v = acc[s][r];
                if constexpr (HAS_ADD) v += addend[idx];
                out[idx] = v;
            }
        }
    }
}

// ---------------------------------------------------------------------------
// Embedding: h[m][d] = x[m] . emb_w[d] + emb_b[d]   (IN_DIM = 4)
// ---------------------------------------------------------------------------
__global__ void embed_kernel(const float* __restrict__ x, const float* __restrict__ ew,
                             const float* __restrict__ eb, float* __restrict__ h) {
    int id = blockIdx.x * blockDim.x + threadIdx.x;
    if (id >= MTOK * DMODEL) return;
    int d = id & (DMODEL - 1);
    int m = id >> 7;
    float acc = eb[d];
    #pragma unroll
    for (int k = 0; k < 4; ++k) acc += x[(size_t)m * 4 + k] * ew[d * 4 + k];
    h[id] = acc;
}

__global__ void resadd_kernel(const float* __restrict__ h, float* __restrict__ resid, int first) {
    int id = blockIdx.x * blockDim.x + threadIdx.x;
    if (id >= MTOK * DMODEL) return;
    resid[id] = first ? h[id] : (h[id] + resid[id]);
}

// ---------------------------------------------------------------------------
// LayerNorm over 128 features; one wave32 per row, 4 elements per lane.
// ---------------------------------------------------------------------------
__global__ __launch_bounds__(256)
void ln_kernel(const float* __restrict__ in, float* __restrict__ out,
               const float* __restrict__ w, const float* __restrict__ b, int rows) {
    int warp = (blockIdx.x * blockDim.x + threadIdx.x) >> 5;
    int lane = threadIdx.x & 31;
    if (warp >= rows) return;
    const float* r = in + (size_t)warp * DMODEL;
    float v[4], s = 0.f, s2 = 0.f;
    #pragma unroll
    for (int i = 0; i < 4; ++i) { v[i] = r[lane + 32 * i]; s += v[i]; s2 += v[i] * v[i]; }
    #pragma unroll
    for (int msk = 16; msk >= 1; msk >>= 1) {
        s  += __shfl_xor(s,  msk, 32);
        s2 += __shfl_xor(s2, msk, 32);
    }
    float mean = s * (1.f / DMODEL);
    float var  = s2 * (1.f / DMODEL) - mean * mean;
    float rstd = rsqrtf(var + 1e-5f);
    float* o = out + (size_t)warp * DMODEL;
    #pragma unroll
    for (int i = 0; i < 4; ++i) {
        int c = lane + 32 * i;
        o[c] = (v[i] - mean) * rstd * w[c] + b[c];
    }
}

// ---------------------------------------------------------------------------
// Causal depthwise conv (K=4) + SiLU.  xz: (M, 512), first 256 cols are xi.
// ---------------------------------------------------------------------------
__global__ void conv_silu_kernel(const float* __restrict__ xz, const float* __restrict__ cw,
                                 const float* __restrict__ cb, float* __restrict__ xi) {
    int id = blockIdx.x * blockDim.x + threadIdx.x;
    if (id >= MTOK * DINNER) return;
    int d = id & (DINNER - 1);
    int m = id >> 8;
    int t = m & (LSEQ - 1);
    float acc = cb[d];
    #pragma unroll
    for (int j = 0; j < 4; ++j) {
        int tt = t - 3 + j;
        if (tt >= 0) acc += cw[d * 4 + j] * xz[(size_t)(m - 3 + j) * (2 * DINNER) + d];
    }
    xi[id] = siluf(acc);
}

// ---------------------------------------------------------------------------
// dt = softplus(dt_raw @ dt_w^T + dt_b)   (K = DT_RANK = 8, too small for WMMA)
// ---------------------------------------------------------------------------
__global__ void dt_kernel(const float* __restrict__ xdbl, const float* __restrict__ dtw,
                          const float* __restrict__ dtb, float* __restrict__ dt) {
    int id = blockIdx.x * blockDim.x + threadIdx.x;
    if (id >= MTOK * DINNER) return;
    int d = id & (DINNER - 1);
    int m = id >> 8;
    float a = dtb[d];
    #pragma unroll
    for (int r = 0; r < DTRANK; ++r) a += xdbl[(size_t)m * XDBLW + r] * dtw[d * DTRANK + r];
    dt[id] = (a > 20.f) ? a : log1pf(__expf(a));
}

// ---------------------------------------------------------------------------
// Selective scan. One block per batch (4 blocks, 1024 threads).
// Thread t owns channel d = t/4 and states n in [32*(t&3), +32), kept in VGPRs.
// B_t / C_t staged in LDS per step; y reduced over the 4 sibling lanes.
// Fused epilogue: y = (sum h*C + u*D) * silu(z).
// ---------------------------------------------------------------------------
__global__ __launch_bounds__(1024)
void scan_kernel(const float* __restrict__ u, const float* __restrict__ dt,
                 const float* __restrict__ xdbl, const float* __restrict__ Alog,
                 const float* __restrict__ Dp, const float* __restrict__ xz,
                 float* __restrict__ yout) {
    int b   = blockIdx.x;
    int tid = threadIdx.x;
    int d   = tid >> 2;
    int q   = tid & 3;
    int n0  = q * 32;

    __shared__ float sB[DSTATE];
    __shared__ float sC[DSTATE];

    float h[32], eA[32];
    #pragma unroll
    for (int j = 0; j < 32; ++j) {
        h[j]  = 0.f;
        eA[j] = __expf(Alog[(size_t)d * DSTATE + n0 + j]);   // A = -exp(A_log)
    }
    float Dd = Dp[d];

    for (int t = 0; t < LSEQ; ++t) {
        int m = b * LSEQ + t;
        if (tid < DSTATE)                sB[tid]          = xdbl[(size_t)m * XDBLW + DTRANK + tid];
        else if (tid < 2 * DSTATE)       sC[tid - DSTATE] = xdbl[(size_t)m * XDBLW + DTRANK + DSTATE + (tid - DSTATE)];
        __syncthreads();

        float dtv = dt[(size_t)m * DINNER + d];
        float uv  = u[(size_t)m * DINNER + d];
        float du  = dtv * uv;
        float acc = 0.f;
        #pragma unroll
        for (int j = 0; j < 32; ++j) {
            float dA = __expf(-dtv * eA[j]);          // exp(dt * A)
            h[j] = dA * h[j] + du * sB[n0 + j];       // h = dA*h + dt*B*u
            acc += h[j] * sC[n0 + j];                 // y += h*C
        }
        acc += __shfl_xor(acc, 1, 32);
        acc += __shfl_xor(acc, 2, 32);
        if (q == 0) {
            float zv = xz[(size_t)m * (2 * DINNER) + DINNER + d];
            float y  = acc + uv * Dd;
            yout[(size_t)m * DINNER + d] = y * siluf(zv);
        }
        __syncthreads();
    }
}

// ---------------------------------------------------------------------------
// GLU: glu[m][j] = mlpy[m][j] * silu(mlpy[m][384+j])
// ---------------------------------------------------------------------------
__global__ void glu_kernel(const float* __restrict__ mlpy, float* __restrict__ glu) {
    int id = blockIdx.x * blockDim.x + threadIdx.x;
    if (id >= MTOK * DHID) return;
    int m = id / DHID;
    int j = id - m * DHID;
    float a = mlpy[(size_t)m * (2 * DHID) + j];
    float g = mlpy[(size_t)m * (2 * DHID) + DHID + j];
    glu[id] = a * siluf(g);
}

// ---------------------------------------------------------------------------
// Head: out[b] = h[b, L-1, :] . head_w + head_b
// ---------------------------------------------------------------------------
__global__ void head_kernel(const float* __restrict__ h, const float* __restrict__ hw,
                            const float* __restrict__ hb, float* __restrict__ out) {
    int b   = blockIdx.x;
    int tid = threadIdx.x;           // 128 threads
    __shared__ float red[DMODEL];
    red[tid] = h[(size_t)(b * LSEQ + LSEQ - 1) * DMODEL + tid] * hw[tid];
    __syncthreads();
    for (int s = 64; s >= 1; s >>= 1) {
        if (tid < s) red[tid] += red[tid + s];
        __syncthreads();
    }
    if (tid == 0) out[b] = red[0] + hb[0];
}

// ---------------------------------------------------------------------------

static inline void launch_gemm(const float* A, const float* W, const float* addend,
                               float* out, int N, int K, hipStream_t s) {
    int Mt = MTOK / 16;
    int Nt = (N + 15) / 16;
    int NGroups = (Nt + 3) / 4;                // 4 N-tiles (64 cols) per wave
    int waves = Mt * NGroups;
    int blocks = (waves + 7) / 8;              // 8 waves (256 threads) per block
    if (addend)
        gemm_f16wmma<true><<<blocks, 256, 0, s>>>(A, W, addend, out, Mt, NGroups, N, K);
    else
        gemm_f16wmma<false><<<blocks, 256, 0, s>>>(A, W, nullptr, out, Mt, NGroups, N, K);
}

extern "C" void kernel_launch(void* const* d_in, const int* in_sizes, int n_in,
                              void* d_out, int out_size, void* d_ws, size_t ws_size,
                              hipStream_t stream) {
    const float* x         = (const float*)d_in[0];
    const float* emb_w     = (const float*)d_in[1];
    const float* emb_b     = (const float*)d_in[2];
    const float* norm1_w   = (const float*)d_in[3];
    const float* norm1_b   = (const float*)d_in[4];
    const float* in_proj_w = (const float*)d_in[5];
    const float* conv_w    = (const float*)d_in[6];
    const float* conv_b    = (const float*)d_in[7];
    const float* x_proj_w  = (const float*)d_in[8];
    const float* dt_proj_w = (const float*)d_in[9];
    const float* dt_proj_b = (const float*)d_in[10];
    const float* A_log     = (const float*)d_in[11];
    const float* D_param   = (const float*)d_in[12];
    const float* out_proj_w= (const float*)d_in[13];
    const float* norm2_w   = (const float*)d_in[14];
    const float* norm2_b   = (const float*)d_in[15];
    const float* fc1_w     = (const float*)d_in[16];
    const float* fc2_w     = (const float*)d_in[17];
    const float* head_w    = (const float*)d_in[18];
    const float* head_b    = (const float*)d_in[19];
    float* out = (float*)d_out;

    float* ws = (float*)d_ws;
    size_t o = 0;
    float* resid = ws + o; o += (size_t)MTOK * DMODEL;
    float* hn    = ws + o; o += (size_t)MTOK * DMODEL;
    float* xz    = ws + o; o += (size_t)MTOK * 2 * DINNER;
    float* xi    = ws + o; o += (size_t)MTOK * DINNER;
    float* xdbl  = ws + o; o += (size_t)MTOK * XDBLW;
    float* dtb   = ws + o; o += (size_t)MTOK * DINNER;
    float* ysc   = ws + o; o += (size_t)MTOK * DINNER;
    float* hbuf  = ws + o; o += (size_t)MTOK * DMODEL;
    float* mlpy  = ws + o; o += (size_t)MTOK * 2 * DHID;
    float* glu   = ws + o; o += (size_t)MTOK * DHID;
    (void)o; (void)ws_size; (void)in_sizes; (void)n_in; (void)out_size;

    embed_kernel<<<(MTOK * DMODEL) / 256, 256, 0, stream>>>(x, emb_w, emb_b, hbuf);

    for (int i = 0; i < 4; ++i) {
        // residual = h (+ residual)
        resadd_kernel<<<(MTOK * DMODEL) / 256, 256, 0, stream>>>(hbuf, resid, i == 0);
        // hn = LN1(residual)
        ln_kernel<<<MTOK / 8, 256, 0, stream>>>(resid, hn,
                                                norm1_w + i * DMODEL, norm1_b + i * DMODEL, MTOK);
        // xz = hn @ in_proj_w^T     (N=512, K=128)
        launch_gemm(hn, in_proj_w + (size_t)i * 2 * DINNER * DMODEL, nullptr, xz,
                    2 * DINNER, DMODEL, stream);
        // xi = silu(causal_dw_conv(xz[:, :256]))
        conv_silu_kernel<<<(MTOK * DINNER) / 256, 256, 0, stream>>>(
            xz, conv_w + (size_t)i * DINNER * 4, conv_b + (size_t)i * DINNER, xi);
        // x_dbl = xi @ x_proj_w^T   (N=264, K=256)
        launch_gemm(xi, x_proj_w + (size_t)i * XDBLW * DINNER, nullptr, xdbl,
                    XDBLW, DINNER, stream);
        // dt = softplus(dt_raw @ dt_w^T + dt_b)
        dt_kernel<<<(MTOK * DINNER) / 256, 256, 0, stream>>>(
            xdbl, dt_proj_w + (size_t)i * DINNER * DTRANK, dt_proj_b + (size_t)i * DINNER, dtb);
        // selective scan (state in VGPRs), fused u*D + silu(z) gate
        scan_kernel<<<4, 1024, 0, stream>>>(xi, dtb, xdbl,
                                            A_log + (size_t)i * DINNER * DSTATE,
                                            D_param + (size_t)i * DINNER, xz, ysc);
        // residual = yscan @ out_proj_w^T + residual   (N=128, K=256)
        launch_gemm(ysc, out_proj_w + (size_t)i * DMODEL * DINNER, resid, resid,
                    DMODEL, DINNER, stream);
        // hn = LN2(residual)
        ln_kernel<<<MTOK / 8, 256, 0, stream>>>(resid, hn,
                                                norm2_w + i * DMODEL, norm2_b + i * DMODEL, MTOK);
        // mlpy = hn @ fc1_w^T       (N=768, K=128)
        launch_gemm(hn, fc1_w + (size_t)i * 2 * DHID * DMODEL, nullptr, mlpy,
                    2 * DHID, DMODEL, stream);
        // glu = mlpy[:, :384] * silu(mlpy[:, 384:])
        glu_kernel<<<(MTOK * DHID) / 256, 256, 0, stream>>>(mlpy, glu);
        // h = glu @ fc2_w^T         (N=128, K=384)
        launch_gemm(glu, fc2_w + (size_t)i * DMODEL * DHID, nullptr, hbuf,
                    DMODEL, DHID, stream);
    }

    head_kernel<<<4, 128, 0, stream>>>(hbuf, head_w, head_b, out);
}
